// AttentionHead_48473000902967
// MI455X (gfx1250) — matrix operand order repaired and checked by reading, not compiled, not run on hardware
//
#include <hip/hip_runtime.h>

#define B_ 256
#define T_ 256
#define C_ 1024
#define D_ 64

typedef __attribute__((ext_vector_type(16))) __bf16 v16bf;
typedef __attribute__((ext_vector_type(8)))  float v8f;
typedef __attribute__((ext_vector_type(8)))  unsigned short u16x8;
typedef __attribute__((ext_vector_type(4)))  float f32x4;
typedef __attribute__((ext_vector_type(4)))  unsigned u32x4;

typedef __attribute__((address_space(3))) const unsigned short lds_cus;

union Frag { u16x8 h[2]; v16bf bf; };

// Pack two f32 into one dword of two bf16 (truncation) with a single v_perm_b32.
__device__ __forceinline__ unsigned pack2bf(float lo, float hi) {
  return __builtin_amdgcn_perm(__float_as_uint(hi), __float_as_uint(lo),
                               0x07060302u);
}

__device__ __forceinline__ v8f wmma_bf16(const Frag& a, const Frag& b, v8f c) {
  return __builtin_amdgcn_wmma_f32_16x16x32_bf16(false, a.bf, false, b.bf,
                                                 (short)0, c, false, false);
}

// Async global->LDS copy (16B per lane), tracked by ASYNCcnt.
__device__ __forceinline__ void async_load_b128(const void* lds_ptr,
                                                const void* gptr) {
  unsigned lds_off = (unsigned)(unsigned long long)(lds_cus*)lds_ptr;
  asm volatile("global_load_async_to_lds_b128 %0, %1, off"
               :: "v"(lds_off), "v"(gptr)
               : "memory");
}
__device__ __forceinline__ void wait_async0() {
  asm volatile("s_wait_asynccnt 0" ::: "memory");
}

// ---------------------------------------------------------------------------
// Kernel 1: fused q/k/v projection. [65536 x 1024] x [1024 x (64|64|64)],
// bf16 WMMA, f32 accum. WG = 256 threads (8 waves), 256 rows/WG, 32 rows/wave
// (2 M-tiles): every staged B fragment feeds two WMMAs, B ring-prefetched two
// tiles deep. The 1024^-0.5 attention scale is folded into q here.
// ---------------------------------------------------------------------------
#define KC   32   // K chunk (WMMA K for bf16)
#define ROWS 256  // rows per workgroup (8 waves x 32)
#define NC   192  // q|k|v fused output columns
#define XPAD 40   // padded row stride (halfs) for x tile  (conflict-free, 16B ok)
#define WPAD 40   // padded col stride (halfs) for W tile (stored transposed)

__global__ __launch_bounds__(256) void proj_qkv(
    const float* __restrict__ x,  const float* __restrict__ Wq,
    const float* __restrict__ Wk, const float* __restrict__ Wv,
    unsigned short* __restrict__ q, unsigned short* __restrict__ k,
    unsigned short* __restrict__ v)
{
  __shared__ __align__(16) unsigned short xs[ROWS * XPAD];  // [row][k]
  __shared__ __align__(16) unsigned short ws[NC * WPAD];    // [n][k] (transposed)

  const int tid  = threadIdx.x;
  const int wave = tid >> 5;
  const int lane = tid & 31;
  const int hi   = lane >> 4;
  const int lo   = lane & 15;
  const long rowbase = (long)blockIdx.x * ROWS;

  v8f acc[24];   // [nt][mtile]: acc[2*nt+mt]
  #pragma unroll
  for (int i = 0; i < 24; ++i) acc[i] = v8f{0.f,0.f,0.f,0.f,0.f,0.f,0.f,0.f};

  // division-free W staging coordinates
  const int n64    = tid & 63;
  const int kpbase = tid >> 6;          // 0..3

  for (int kc = 0; kc < C_; kc += KC) {
    // stage x chunk: 256 rows x 32 k -> bf16. One row per thread: 128B of
    // contiguous NT f32 loads, pack via v_perm, four b128 LDS stores.
    {
      const f32x4* src = (const f32x4*)(x + (rowbase + tid) * C_ + kc);
      u32x4* dst = (u32x4*)(xs + tid * XPAD);
      #pragma unroll
      for (int g = 0; g < 4; ++g) {
        f32x4 fa = __builtin_nontemporal_load(src + 2 * g);
        f32x4 fb = __builtin_nontemporal_load(src + 2 * g + 1);
        u32x4 pkd;
        pkd[0] = pack2bf(fa[0], fa[1]); pkd[1] = pack2bf(fa[2], fa[3]);
        pkd[2] = pack2bf(fb[0], fb[1]); pkd[3] = pack2bf(fb[2], fb[3]);
        dst[g] = pkd;
      }
    }
    // stage W chunk transposed: ws[n][k]; two k per thread -> one b32 store
    #pragma unroll
    for (int mat = 0; mat < 3; ++mat) {
      const float* Wsrc = (mat == 0) ? Wq : (mat == 1) ? Wk : Wv;
      #pragma unroll
      for (int j = 0; j < 4; ++j) {
        int kp = kpbase + 4 * j;        // 0..15
        const float* p = Wsrc + (long)(kc + 2 * kp) * D_ + n64;
        *(unsigned*)(ws + (mat * 64 + n64) * WPAD + 2 * kp) =
            pack2bf(p[0], p[D_]);
      }
    }
    __syncthreads();

    // Two A fragments (16x32 bf16 each): lane = row, K = kb+j / kb+8+j
    Frag a0, a1;
    {
      int row = wave * 32 + lo;
      int kb  = hi * 16;
      const unsigned short* p = xs + row * XPAD + kb;
      a0.h[0] = *(const u16x8*)(p);
      a0.h[1] = *(const u16x8*)(p + 8);
      p += 16 * XPAD;
      a1.h[0] = *(const u16x8*)(p);
      a1.h[1] = *(const u16x8*)(p + 8);
    }
    // B ring buffer, prefetched two N-tiles deep
    const int kb2 = hi * 16;
    Frag bb[2];
    {
      const unsigned short* p = ws + lo * WPAD + kb2;
      bb[0].h[0] = *(const u16x8*)(p);
      bb[0].h[1] = *(const u16x8*)(p + 8);
      p = ws + (16 + lo) * WPAD + kb2;
      bb[1].h[0] = *(const u16x8*)(p);
      bb[1].h[1] = *(const u16x8*)(p + 8);
    }
    #pragma unroll
    for (int nt = 0; nt < 12; ++nt) {
      Frag& bc = bb[nt & 1];
      acc[2 * nt]     = wmma_bf16(a0, bc, acc[2 * nt]);
      acc[2 * nt + 1] = wmma_bf16(a1, bc, acc[2 * nt + 1]);
      if (nt + 2 < 12) {
        const unsigned short* p = ws + ((nt + 2) * 16 + lo) * WPAD + kb2;
        bc.h[0] = *(const u16x8*)(p);
        bc.h[1] = *(const u16x8*)(p + 8);
      }
    }
    __syncthreads();
  }

  // C/D layout: VGPR r -> row (r + 8*hi), lane%16 -> col.
  // q gets the 1024^-0.5 = 1/32 attention scale folded in here.
  #pragma unroll
  for (int nt = 0; nt < 12; ++nt) {
    unsigned short* outp = (nt < 4) ? q : (nt < 8) ? k : v;
    float scale = (nt < 4) ? 0.03125f : 1.0f;
    int col = (nt & 3) * 16 + lo;
    #pragma unroll
    for (int mt = 0; mt < 2; ++mt) {
      long base = rowbase + wave * 32 + mt * 16 + hi * 8;
      #pragma unroll
      for (int r = 0; r < 8; ++r) {
        // (trunc (srl x 16)) -> global_store_d16_hi_b16, zero extra VALU
        unsigned bits = __float_as_uint(acc[2 * nt + mt][r] * scale);
        outp[(base + r) * D_ + col] = (unsigned short)(bits >> 16);
      }
    }
  }
}

// ---------------------------------------------------------------------------
// Kernel 2: causal flash attention over bf16 q/k/v, f32 accum + online softmax.
// WG = 256 threads (8 waves); wave owns 16 query rows; WG owns 128 rows of one
// batch; k tile staged via GLOBAL_LOAD_ASYNC_TO_LDS (ASYNCcnt), v transposed
// manually; both shared by all 8 waves.
// ---------------------------------------------------------------------------
#define KSTR 72   // ks row stride (halfs): [s][d], conflict-free, 16B-aligned
#define VSTR 40   // vT row stride (halfs): [d][s]
#define PSTR 40   // p scratch row stride (halfs): [m][s]

__global__ __launch_bounds__(256) void attn_kernel(
    const unsigned short* __restrict__ q, const unsigned short* __restrict__ k,
    const unsigned short* __restrict__ v, float* __restrict__ out)
{
  __shared__ __align__(16) unsigned short ks[32 * KSTR];     // keys [s][d]
  __shared__ __align__(16) unsigned short vT[D_ * VSTR];     // values [d][s]
  __shared__ __align__(16) unsigned short ps[8][16 * PSTR];  // per-wave probs [m][s]

  const int tid  = threadIdx.x;
  const int wave = tid >> 5;
  const int lane = tid & 31;
  const int hi   = lane >> 4;
  const int lo   = lane & 15;
  const int b    = blockIdx.x >> 1;
  const int tile = blockIdx.x & 1;
  const int qbase = tile * 128 + wave * 16;
  const long bofs = (long)b * T_ * D_;

  // q A-fragments for D=64 split into two K=32 windows (scale pre-folded)
  Frag aq[2];
  {
    int row = qbase + lo;
    int kb  = hi * 8;
    const unsigned short* p0 = q + bofs + (long)row * D_ + kb;
    aq[0].h[0] = *(const u16x8*)(p0);
    aq[0].h[1] = *(const u16x8*)(p0 + 16);
    aq[1].h[0] = *(const u16x8*)(p0 + 32);
    aq[1].h[1] = *(const u16x8*)(p0 + 48);
  }

  v8f acc[4];
  #pragma unroll
  for (int i = 0; i < 4; ++i) acc[i] = v8f{0.f,0.f,0.f,0.f,0.f,0.f,0.f,0.f};
  float m[8], l[8];
  #pragma unroll
  for (int r = 0; r < 8; ++r) { m[r] = -INFINITY; l[r] = 0.f; }

  const int smax    = qbase + 15;
  const int nchunks = (tile * 128 + 128) / 32;

  for (int c = 0; c < nchunks; ++c) {
    const int sc = c * 32;
    // stage 32 keys [s][d] (async, no VGPR round-trip) + 32 values transposed
    {
      int s  = tid >> 3;
      int d0 = (tid & 7) * 8;
      async_load_b128(ks + s * KSTR + d0,
                      k + bofs + (long)(sc + s) * D_ + d0);
      u16x8 vv = *(const u16x8*)(v + bofs + (long)(sc + s) * D_ + d0);
      #pragma unroll
      for (int j = 0; j < 8; ++j) vT[(d0 + j) * VSTR + s] = vv[j];
      wait_async0();
    }
    __syncthreads();

    if (sc <= smax) {
      // ---- scores: q[16x64] @ k^T[64x32] -> two 16x16 f32 tiles
      v8f s0 = v8f{0.f,0.f,0.f,0.f,0.f,0.f,0.f,0.f};
      v8f s1 = s0;
      #pragma unroll
      for (int dc = 0; dc < 2; ++dc) {
        int kb2 = hi * 8;
        const unsigned short* p0 = ks + lo * KSTR + dc * 32 + kb2;
        Frag b0, b1;
        b0.h[0] = *(const u16x8*)(p0);
        b0.h[1] = *(const u16x8*)(p0 + 16);
        const unsigned short* p1 = p0 + 16 * KSTR;
        b1.h[0] = *(const u16x8*)(p1);
        b1.h[1] = *(const u16x8*)(p1 + 16);
        s0 = wmma_bf16(aq[dc], b0, s0);
        s1 = wmma_bf16(aq[dc], b1, s1);
      }

      // ---- mask + online softmax (row = r + 8*hi across VGPRs, col = lo)
      #pragma unroll
      for (int r = 0; r < 8; ++r) {
        int qrow = qbase + r + 8 * hi;
        float e0 = (sc + lo      <= qrow) ? s0[r] : -INFINITY;
        float e1 = (sc + 16 + lo <= qrow) ? s1[r] : -INFINITY;
        float mx = fmaxf(e0, e1);
        mx = fmaxf(mx, __shfl_xor(mx, 1));
        mx = fmaxf(mx, __shfl_xor(mx, 2));
        mx = fmaxf(mx, __shfl_xor(mx, 4));
        mx = fmaxf(mx, __shfl_xor(mx, 8));
        float mn   = fmaxf(m[r], mx);
        float corr = __expf(m[r] - mn);   // first chunk: exp(-inf)=0
        float p0v  = __expf(e0 - mn);
        float p1v  = __expf(e1 - mn);
        float rs   = p0v + p1v;
        rs += __shfl_xor(rs, 1);
        rs += __shfl_xor(rs, 2);
        rs += __shfl_xor(rs, 4);
        rs += __shfl_xor(rs, 8);
        l[r] = l[r] * corr + rs;
        m[r] = mn;
        #pragma unroll
        for (int nt = 0; nt < 4; ++nt) acc[nt][r] *= corr;
        // spill probs (C layout) to per-wave LDS scratch for A-layout reload;
        // high-16 store -> ds_store_b16_d16_hi
        unsigned short* pp = ps[wave] + (r + 8 * hi) * PSTR;
        pp[lo]      = (unsigned short)(__float_as_uint(p0v) >> 16);
        pp[16 + lo] = (unsigned short)(__float_as_uint(p1v) >> 16);
      }

      // ---- PV: p[16x32] @ v[32x64], accumulate into acc
      Frag ap;
      {
        const unsigned short* pp = ps[wave] + lo * PSTR + hi * 8;
        ap.h[0] = *(const u16x8*)(pp);
        ap.h[1] = *(const u16x8*)(pp + 16);
      }
      #pragma unroll
      for (int nt = 0; nt < 4; ++nt) {
        Frag bv;
        const unsigned short* pv = vT + (nt * 16 + lo) * VSTR + hi * 8;
        bv.h[0] = *(const u16x8*)(pv);
        bv.h[1] = *(const u16x8*)(pv + 16);
        acc[nt] = wmma_bf16(ap, bv, acc[nt]);
      }
    }
    __syncthreads();
  }

  // normalize + write f32 output (streamed once -> non-temporal)
  #pragma unroll
  for (int nt = 0; nt < 4; ++nt) {
    #pragma unroll
    for (int r = 0; r < 8; ++r) {
      int row = qbase + r + 8 * hi;
      __builtin_nontemporal_store(acc[nt][r] / l[r],
                                  out + bofs + (long)row * D_ + nt * 16 + lo);
    }
  }
}

extern "C" void kernel_launch(void* const* d_in, const int* in_sizes, int n_in,
                              void* d_out, int out_size, void* d_ws, size_t ws_size,
                              hipStream_t stream) {
  (void)in_sizes; (void)n_in; (void)out_size; (void)ws_size;
  const float* x  = (const float*)d_in[0];
  const float* Wq = (const float*)d_in[1];
  const float* Wk = (const float*)d_in[2];
  const float* Wv = (const float*)d_in[3];

  unsigned short* q = (unsigned short*)d_ws;             // bf16 q: 8 MB
  unsigned short* k = q + (size_t)B_ * T_ * D_;          // bf16 k: 8 MB
  unsigned short* v = k + (size_t)B_ * T_ * D_;          // bf16 v: 8 MB

  proj_qkv<<<(B_ * T_) / ROWS, 256, 0, stream>>>(x, Wq, Wk, Wv, q, k, v);
  attn_kernel<<<B_ * 2, 256, 0, stream>>>(q, k, v, (float*)d_out);
}
